// NeuronLlamaMLP_8169027797430
// MI455X (gfx1250) — compile-verified, matrix operand order checked by reading, not run on hardware
//
#include <hip/hip_runtime.h>
#include <hip/hip_bf16.h>

// ---------------------------------------------------------------------------
// Fused RMSNorm + SwiGLU MLP for MI455X (gfx1250, wave32, WMMA bf16 + TDM)
//   x[2048,4096] fp32 -> rmsnorm -> bf16 nx
//   g = nx @ gate_w^T ; u = nx @ up_w^T   (v_wmma_f32_16x16x32_bf16)
//   h = silu(g)*u  (bf16, 2048x14336)
//   out = h @ down_w^T (fp32)
// A tiles move via Tensor Data Mover (tensor_load_to_lds, TENSORcnt),
// triple-buffered; B weight tiles are staged fp32->bf16 through registers
// (v_cvt_pk_bf16_f32), double-buffered; one workgroup barrier per K stage.
// Workspace: nx (16 MiB) + h (56 MiB) = 72 MiB of d_ws.
// ---------------------------------------------------------------------------

typedef __attribute__((ext_vector_type(16))) __bf16 bf16x16;
typedef __attribute__((ext_vector_type(8)))  __bf16 bf16x8;
typedef __attribute__((ext_vector_type(2)))  __bf16 bf16x2;
typedef __attribute__((ext_vector_type(8)))  float  f32x8;
typedef __attribute__((ext_vector_type(4)))  unsigned u32x4;
typedef __attribute__((ext_vector_type(8)))  int     i32x8;
typedef __attribute__((ext_vector_type(4)))  int     i32x4;

#define RMS_EPS 1e-5f
#define S_TOT 2048
#define H_TOT 4096
#define I_TOT 14336

#define BM 128          // block tile M (sequence rows)
#define BN 64           // block tile N (output cols per weight)
#define KS 64           // K depth per stage = 2 WMMA k-steps
#define LDSTR 72        // LDS row stride (bf16): 64 data + 8 pad -> 144B rows
#define A_ELEMS (BM * LDSTR)
#define B_ELEMS (BN * LDSTR)

#if __has_builtin(__builtin_amdgcn_tensor_load_to_lds)
#define HAVE_TDM 1
#else
#define HAVE_TDM 0
#endif

// fp32 pair -> packed bf16x2 dword: lowers to a single v_cvt_pk_bf16_f32.
__device__ __forceinline__ unsigned bfpair(float lo, float hi) {
  bf16x2 p;
  p[0] = (__bf16)lo;
  p[1] = (__bf16)hi;
  return __builtin_bit_cast(unsigned, p);
}
__device__ __forceinline__ uint4 pack8(float4 a, float4 b) {
  return make_uint4(bfpair(a.x, a.y), bfpair(a.z, a.w),
                    bfpair(b.x, b.y), bfpair(b.z, b.w));
}

// One 16x32 bf16 WMMA operand fragment from an LDS tile (two ds_load_b128).
__device__ __forceinline__ bf16x16 load_frag(const __bf16* tile, int row, int cb) {
  union { bf16x16 v; bf16x8 h[2]; } f;
  const __bf16* p = tile + row * LDSTR + cb;
  f.h[0] = *(const bf16x8*)(p);
  f.h[1] = *(const bf16x8*)(p + 16);
  return f.v;
}

#if HAVE_TDM
// Issue a TDM 2D tile load: BM x KS bf16 tile from a row-major [rows, k_total]
// bf16 tensor into LDS with 16B padding every 128B row (-> LDSTR stride).
__device__ __forceinline__ void tdm_load_2d(const void* gsrc, unsigned lds_off,
                                            int k_total, int rows_total) {
  unsigned long long ga = (unsigned long long)(size_t)gsrc;
  u32x4 g0;
  g0[0] = 1u;                                  // count=1, user descriptor
  g0[1] = lds_off;                             // lds_addr (bytes)
  g0[2] = (unsigned)ga;                        // global_addr[31:0]
  g0[3] = (unsigned)((ga >> 32) & 0x1FFFFFFull) | (2u << 30);  // [56:32]|type=2
  i32x8 g1;
  g1[0] = (1 << 16)        // data_size = 2 bytes
        | (1 << 20)        // pad_enable
        | (4 << 22)        // pad_interval: 32 DWORDs (one 128B row)
        | (3 << 25);       // pad_amount: 4 DWORDs (16B)
  g1[1] = (k_total & 0xFFFF) << 16;                              // tensor_dim0 lo
  g1[2] = ((k_total >> 16) & 0xFFFF) | ((rows_total & 0xFFFF) << 16);
  g1[3] = ((rows_total >> 16) & 0xFFFF) | (KS << 16);            // tile_dim0=64
  g1[4] = BM;                                  // tile_dim1=128, tile_dim2=0
  g1[5] = k_total;                             // tensor_dim0_stride lo32
  g1[6] = 0;
  g1[7] = 0;
  i32x4 z4 = {0, 0, 0, 0};
  i32x8 z8 = {0, 0, 0, 0, 0, 0, 0, 0};
  __builtin_amdgcn_tensor_load_to_lds(g0, g1, z4, z4, z8, 0);
}
#endif

// ---------------------------------------------------------------------------
// Kernel 1: RMSNorm -> bf16.  One block (256 thr) per sequence row.
// ---------------------------------------------------------------------------
__global__ __launch_bounds__(256)
void rmsnorm_bf16_kernel(const float* __restrict__ x,
                         const float* __restrict__ ln_w,
                         __bf16* __restrict__ nx) {
  const int row = blockIdx.x;
  const int tid = threadIdx.x;
  const float* xr = x + (size_t)row * H_TOT;

  __shared__ float red[256];

  float4 v[4];
  const float4* xr4 = (const float4*)xr + tid * 4;
  float ss = 0.f;
#pragma unroll
  for (int i = 0; i < 4; ++i) {
    v[i] = xr4[i];
    ss += v[i].x * v[i].x + v[i].y * v[i].y + v[i].z * v[i].z + v[i].w * v[i].w;
  }
  red[tid] = ss;
  __syncthreads();
#pragma unroll
  for (int off = 128; off > 0; off >>= 1) {
    if (tid < off) red[tid] += red[tid + off];
    __syncthreads();
  }
  const float scale = rsqrtf(red[0] * (1.0f / (float)H_TOT) + RMS_EPS);

  const float4* w4 = (const float4*)ln_w + tid * 4;
  unsigned o[8];
#pragma unroll
  for (int i = 0; i < 4; ++i) {
    float4 w = w4[i];
    o[i * 2 + 0] = bfpair(v[i].x * scale * w.x, v[i].y * scale * w.y);
    o[i * 2 + 1] = bfpair(v[i].z * scale * w.z, v[i].w * scale * w.w);
  }
  uint4* dst = (uint4*)(nx + (size_t)row * H_TOT) + tid * 2;
  dst[0] = make_uint4(o[0], o[1], o[2], o[3]);
  dst[1] = make_uint4(o[4], o[5], o[6], o[7]);
}

// ---------------------------------------------------------------------------
// Kernel 2: fused gate+up GEMM + SwiGLU epilogue -> bf16 h buffer.
// 256 threads = 8 wave32 (4 M x 2 N), wave tile 32x32 per weight.
// ---------------------------------------------------------------------------
__global__ __launch_bounds__(256)
void gateup_kernel(const __bf16* __restrict__ nx,
                   const float* __restrict__ gate_w,
                   const float* __restrict__ up_w,
                   __bf16* __restrict__ hbuf) {
  __shared__ __bf16 As[3 * A_ELEMS];
  __shared__ __bf16 Bg[2 * B_ELEMS];
  __shared__ __bf16 Bu[2 * B_ELEMS];

  const int tid  = threadIdx.x;
  const int lane = tid & 31;
  const int wave = tid >> 5;
  const int wm   = wave & 3;
  const int wn   = wave >> 2;
  const int r    = lane & 15;
  const int half = lane >> 4;

  const int s0 = blockIdx.y * BM;
  const int n0 = blockIdx.x * BN;

  // B staging: 64x64 fp32 tile, 2 chunks of 8 floats per thread.
  const int br0 = tid >> 3;
  const int bc0 = (tid & 7) * 8;
  const int br1 = (tid + 256) >> 3;
  const int bc1 = ((tid + 256) & 7) * 8;
  const float* gw0 = gate_w + (size_t)(n0 + br0) * H_TOT + bc0;
  const float* gw1 = gate_w + (size_t)(n0 + br1) * H_TOT + bc1;
  const float* uw0 = up_w   + (size_t)(n0 + br0) * H_TOT + bc0;
  const float* uw1 = up_w   + (size_t)(n0 + br1) * H_TOT + bc1;

  uint4 rg0, rg1, ru0, ru1;
  auto loadB = [&](int k0) {
    const float* p;
    p = gw0 + k0; rg0 = pack8(((const float4*)p)[0], ((const float4*)p)[1]);
    p = gw1 + k0; rg1 = pack8(((const float4*)p)[0], ((const float4*)p)[1]);
    p = uw0 + k0; ru0 = pack8(((const float4*)p)[0], ((const float4*)p)[1]);
    p = uw1 + k0; ru1 = pack8(((const float4*)p)[0], ((const float4*)p)[1]);
  };

#if HAVE_TDM
  const unsigned lds_a0 = (unsigned)(size_t)(void*)&As[0];
#else
  uint4 ra[4];
  auto loadA = [&](int k0) {
#pragma unroll
    for (int j = 0; j < 4; ++j) {
      int c = tid + j * 256, row = c >> 3, col = (c & 7) * 8;
      ra[j] = *(const uint4*)(nx + (size_t)(s0 + row) * H_TOT + k0 + col);
    }
  };
  auto storeA = [&](__bf16* dst) {
#pragma unroll
    for (int j = 0; j < 4; ++j) {
      int c = tid + j * 256, row = c >> 3, col = (c & 7) * 8;
      *(uint4*)(dst + row * LDSTR + col) = ra[j];
    }
  };
#endif

  f32x8 accg[2][2] = {};
  f32x8 accu[2][2] = {};
  const int NS = H_TOT / KS;

#if HAVE_TDM
  if (wave == 0)
    tdm_load_2d(nx + (size_t)s0 * H_TOT, lds_a0, H_TOT, S_TOT);
#else
  loadA(0);
#endif
  loadB(0);

  int abuf = 0;
  for (int t = 0; t < NS; ++t) {
    const int bbuf = t & 1;
    __bf16* bgs = Bg + bbuf * B_ELEMS;
    __bf16* bus = Bu + bbuf * B_ELEMS;
#if !HAVE_TDM
    storeA(As + abuf * A_ELEMS);
#endif
    *(uint4*)(bgs + br0 * LDSTR + bc0) = rg0;
    *(uint4*)(bgs + br1 * LDSTR + bc1) = rg1;
    *(uint4*)(bus + br0 * LDSTR + bc0) = ru0;
    *(uint4*)(bus + br1 * LDSTR + bc1) = ru1;

    const bool more = (t + 1) < NS;
    const int anext = (abuf == 2) ? 0 : abuf + 1;
    if (more) {
      const int k1 = (t + 1) * KS;
#if HAVE_TDM
      if (wave == 0)
        tdm_load_2d(nx + (size_t)s0 * H_TOT + k1,
                    lds_a0 + (unsigned)(anext * A_ELEMS * 2), H_TOT, S_TOT);
#else
      loadA(k1);
#endif
      loadB(k1);
    }
#if HAVE_TDM
    if (wave == 0) {
      if (more) __builtin_amdgcn_s_wait_tensorcnt(1);   // stage t complete
      else      __builtin_amdgcn_s_wait_tensorcnt(0);
    }
#endif
    __syncthreads();

    const __bf16* at = As + abuf * A_ELEMS;
#pragma unroll
    for (int kk = 0; kk < 2; ++kk) {
      const int cb = kk * 32 + half * 8;
      bf16x16 af[2], gf[2], uf[2];
#pragma unroll
      for (int mt = 0; mt < 2; ++mt)
        af[mt] = load_frag(at, wm * 32 + mt * 16 + r, cb);
#pragma unroll
      for (int nt = 0; nt < 2; ++nt) {
        gf[nt] = load_frag(bgs, wn * 32 + nt * 16 + r, cb);
        uf[nt] = load_frag(bus, wn * 32 + nt * 16 + r, cb);
      }
#pragma unroll
      for (int mt = 0; mt < 2; ++mt)
#pragma unroll
        for (int nt = 0; nt < 2; ++nt) {
          accg[mt][nt] = __builtin_amdgcn_wmma_f32_16x16x32_bf16(
              false, af[mt], false, gf[nt], (short)0, accg[mt][nt], false, false);
          accu[mt][nt] = __builtin_amdgcn_wmma_f32_16x16x32_bf16(
              false, af[mt], false, uf[nt], (short)0, accu[mt][nt], false, false);
        }
    }
    abuf = anext;
  }

  // epilogue: h = silu(g)*u.  C/D layout: M = half*8 + vgpr, N = lane%16.
#pragma unroll
  for (int mt = 0; mt < 2; ++mt)
#pragma unroll
    for (int nt = 0; nt < 2; ++nt) {
      const int col = n0 + wn * 32 + nt * 16 + r;
#pragma unroll
      for (int v = 0; v < 8; ++v) {
        const int srow = s0 + wm * 32 + mt * 16 + half * 8 + v;
        float g = accg[mt][nt][v];
        float u = accu[mt][nt][v];
        hbuf[(size_t)srow * I_TOT + col] = (__bf16)((g / (1.0f + __expf(-g))) * u);
      }
    }
}

// ---------------------------------------------------------------------------
// Kernel 3: down GEMM.  A: h bf16 [S,I], B^T: down_w rows fp32 [H,I].
// ---------------------------------------------------------------------------
__global__ __launch_bounds__(256)
void down_kernel(const __bf16* __restrict__ hbuf,
                 const float* __restrict__ down_w,
                 float* __restrict__ out) {
  __shared__ __bf16 As[3 * A_ELEMS];
  __shared__ __bf16 Bd[2 * B_ELEMS];

  const int tid  = threadIdx.x;
  const int lane = tid & 31;
  const int wave = tid >> 5;
  const int wm   = wave & 3;
  const int wn   = wave >> 2;
  const int r    = lane & 15;
  const int half = lane >> 4;

  const int s0 = blockIdx.y * BM;
  const int n0 = blockIdx.x * BN;

  const int br0 = tid >> 3;
  const int bc0 = (tid & 7) * 8;
  const int br1 = (tid + 256) >> 3;
  const int bc1 = ((tid + 256) & 7) * 8;
  const float* dw0 = down_w + (size_t)(n0 + br0) * I_TOT + bc0;
  const float* dw1 = down_w + (size_t)(n0 + br1) * I_TOT + bc1;

  uint4 rd0, rd1;
  auto loadB = [&](int k0) {
    const float* p;
    p = dw0 + k0; rd0 = pack8(((const float4*)p)[0], ((const float4*)p)[1]);
    p = dw1 + k0; rd1 = pack8(((const float4*)p)[0], ((const float4*)p)[1]);
  };

#if HAVE_TDM
  const unsigned lds_a0 = (unsigned)(size_t)(void*)&As[0];
#else
  uint4 ra[4];
  auto loadA = [&](int k0) {
#pragma unroll
    for (int j = 0; j < 4; ++j) {
      int c = tid + j * 256, row = c >> 3, col = (c & 7) * 8;
      ra[j] = *(const uint4*)(hbuf + (size_t)(s0 + row) * I_TOT + k0 + col);
    }
  };
  auto storeA = [&](__bf16* dst) {
#pragma unroll
    for (int j = 0; j < 4; ++j) {
      int c = tid + j * 256, row = c >> 3, col = (c & 7) * 8;
      *(uint4*)(dst + row * LDSTR + col) = ra[j];
    }
  };
#endif

  f32x8 acc[2][2] = {};
  const int NS = I_TOT / KS;

#if HAVE_TDM
  if (wave == 0)
    tdm_load_2d(hbuf + (size_t)s0 * I_TOT, lds_a0, I_TOT, S_TOT);
#else
  loadA(0);
#endif
  loadB(0);

  int abuf = 0;
  for (int t = 0; t < NS; ++t) {
    const int bbuf = t & 1;
    __bf16* bds = Bd + bbuf * B_ELEMS;
#if !HAVE_TDM
    storeA(As + abuf * A_ELEMS);
#endif
    *(uint4*)(bds + br0 * LDSTR + bc0) = rd0;
    *(uint4*)(bds + br1 * LDSTR + bc1) = rd1;

    const bool more = (t + 1) < NS;
    const int anext = (abuf == 2) ? 0 : abuf + 1;
    if (more) {
      const int k1 = (t + 1) * KS;
#if HAVE_TDM
      if (wave == 0)
        tdm_load_2d(hbuf + (size_t)s0 * I_TOT + k1,
                    lds_a0 + (unsigned)(anext * A_ELEMS * 2), I_TOT, S_TOT);
#else
      loadA(k1);
#endif
      loadB(k1);
    }
#if HAVE_TDM
    if (wave == 0) {
      if (more) __builtin_amdgcn_s_wait_tensorcnt(1);
      else      __builtin_amdgcn_s_wait_tensorcnt(0);
    }
#endif
    __syncthreads();

    const __bf16* at = As + abuf * A_ELEMS;
#pragma unroll
    for (int kk = 0; kk < 2; ++kk) {
      const int cb = kk * 32 + half * 8;
      bf16x16 af[2], bf[2];
#pragma unroll
      for (int mt = 0; mt < 2; ++mt)
        af[mt] = load_frag(at, wm * 32 + mt * 16 + r, cb);
#pragma unroll
      for (int nt = 0; nt < 2; ++nt)
        bf[nt] = load_frag(bds, wn * 32 + nt * 16 + r, cb);
#pragma unroll
      for (int mt = 0; mt < 2; ++mt)
#pragma unroll
        for (int nt = 0; nt < 2; ++nt)
          acc[mt][nt] = __builtin_amdgcn_wmma_f32_16x16x32_bf16(
              false, af[mt], false, bf[nt], (short)0, acc[mt][nt], false, false);
    }
    abuf = anext;
  }

#pragma unroll
  for (int mt = 0; mt < 2; ++mt)
#pragma unroll
    for (int nt = 0; nt < 2; ++nt) {
      const int col = n0 + wn * 32 + nt * 16 + r;
#pragma unroll
      for (int v = 0; v < 8; ++v) {
        const int srow = s0 + wm * 32 + mt * 16 + half * 8 + v;
        out[(size_t)srow * H_TOT + col] = acc[mt][nt][v];
      }
    }
}

// ---------------------------------------------------------------------------
extern "C" void kernel_launch(void* const* d_in, const int* in_sizes, int n_in,
                              void* d_out, int out_size, void* d_ws, size_t ws_size,
                              hipStream_t stream) {
  const float* x      = (const float*)d_in[0];
  const float* ln_w   = (const float*)d_in[1];
  const float* gate_w = (const float*)d_in[2];
  const float* up_w   = (const float*)d_in[3];
  const float* down_w = (const float*)d_in[4];
  float* out = (float*)d_out;

  __bf16* nx   = (__bf16*)d_ws;                                        // 16 MiB
  __bf16* hbuf = (__bf16*)((char*)d_ws + (size_t)S_TOT * H_TOT * 2);   // 56 MiB

  rmsnorm_bf16_kernel<<<dim3(S_TOT), dim3(256), 0, stream>>>(x, ln_w, nx);
  gateup_kernel<<<dim3(I_TOT / BN, S_TOT / BM), dim3(256), 0, stream>>>(nx, gate_w, up_w, hbuf);
  down_kernel<<<dim3(H_TOT / BN, S_TOT / BM), dim3(256), 0, stream>>>(hbuf, down_w, out);
}
